// LayerNormNeuralCDE_850403525423
// MI455X (gfx1250) — matrix operand order repaired and checked by reading, not compiled, not run on hardware
//
#include <hip/hip_runtime.h>
#include <hip/hip_bf16.h>

typedef __bf16 bf16_t;
typedef __attribute__((ext_vector_type(16))) __bf16 v16bf;
typedef __attribute__((ext_vector_type(8)))  float  v8f;

#define WMMA_BF16(a,b,c) __builtin_amdgcn_wmma_f32_16x16x32_bf16(false,(a),false,(b),(short)0,(c),false,false)

// ---- problem dims ----
// B=128, S=32, IN=32, HID=128, 2*HID=256, HID*IN=4096, steps=31

// ---- workspace layout (bytes) ----
static constexpr size_t OFF_W1T  = 0;                          // bf16 [256][128]  vf_W1^T
static constexpr size_t OFF_W2T  = OFF_W1T  + 256*128*2;       // bf16 [256][256]  vf_W2^T
static constexpr size_t OFF_W3T  = OFF_W2T  + 256*256*2;       // bf16 [4096][256] vf_W3^T
static constexpr size_t OFF_EW1T = OFF_W3T  + 4096ull*256*2;   // bf16 [128][32]   enc_W1^T
static constexpr size_t OFF_EW2T = OFF_EW1T + 128*32*2;        // bf16 [128][128]  enc_W2^T
static constexpr size_t OFF_WVT  = OFF_EW2T + 128*128*2;       // bf16 [128][128]  Wv^T
static constexpr size_t OFF_WOT  = OFF_WVT  + 128*128*2;       // bf16 [128][128]  Wo^T
static constexpr size_t OFF_DW1T = OFF_WOT  + 128*128*2;       // bf16 [128][128]  dec_W1^T
static constexpr size_t OFF_DW2T = OFF_DW1T + 128*128*2;       // bf16 [16][128]   dec_W2^T (N padded to 16)
static constexpr size_t OFF_Z    = OFF_DW2T + 16*128*2;        // f32  [128][128]  z state
static constexpr size_t OFF_K    = OFF_Z    + 128*128*4;       // f32  [4][128][128] k1..k4
static constexpr size_t OFF_H1   = OFF_K    + 4ull*128*128*4;  // bf16 [128][256]
static constexpr size_t OFF_H2   = OFF_H1   + 128*256*2;       // bf16 [128][256]
static constexpr size_t OFF_DXS  = OFF_H2   + 128*256*2;       // f32  [31][128][32] dX/dt
static constexpr size_t OFF_HB   = OFF_DXS  + 31ull*128*32*4;  // f32  [31][128]     h step
static constexpr size_t OFF_BK   = OFF_HB   + 31ull*128*4;     // f32  [31][128][128] b3 (.) dxdt

__device__ inline float gelu_f(float x) {
    return 0.5f * x * (1.0f + erff(x * 0.70710678118654752f));
}

// CDNA5 async memory->LDS copy (16B per lane), tracked by ASYNCcnt.
// LDS operand: workgroup-relative LDS byte offset = low 32 bits of generic shared address.
__device__ inline void async_copy_b128(const void* gptr, void* lds_ptr) {
    unsigned lds_off = (unsigned)(uintptr_t)lds_ptr;
    unsigned long long gaddr = (unsigned long long)(uintptr_t)gptr;
    asm volatile("global_load_async_to_lds_b128 %0, %1, off"
                 :: "v"(lds_off), "v"(gaddr) : "memory");
}
__device__ inline void wait_async0() {
    asm volatile("s_wait_asynccnt 0x0" ::: "memory");
}

// Butterfly-XOR lane exchange via single ds_swizzle_b32 (group-of-32 bit mode:
// offset = and_mask 0x1F | or_mask 0<<5 | xor_mask<<10). mask<16 stays in 16-lane half.
template <int MASK>
__device__ inline float swz_add(float v) {
    int x = __builtin_amdgcn_ds_swizzle(__float_as_int(v), (MASK << 10) | 0x1F);
    return v + __int_as_float(x);
}

// A operand (16x32 bf16 tile, row-major source):
// lanes 0-15: M=lane, K chunks {0..7, 16..23}; lanes 16-31: M=lane-16, K {8..15, 24..31}
__device__ inline v16bf ldA(const bf16_t* src, int ld, int row0, int k0) {
    int lane = threadIdx.x & 31;
    int sub  = lane >> 4;
    int m    = lane & 15;
    const bf16_t* p = src + (size_t)(row0 + m) * ld + k0 + sub * 8;
    v16bf a;
#pragma unroll
    for (int e = 0; e < 8; ++e) { a[e] = p[e]; a[e + 8] = p[e + 16]; }
    return a;
}

// B operand (32x16 bf16 tile) gathered from transposed weights Wt[N][K]:
// lanes 0-15: N=lane, K chunks {0..7, 16..23}; lanes 16-31: N=lane-16, K {8..15, 24..31}
__device__ inline v16bf ldB(const bf16_t* wt, int ld, int n0, int k0) {
    int lane = threadIdx.x & 31;
    int sub  = lane >> 4;
    int n    = lane & 15;
    const bf16_t* p = wt + (size_t)(n0 + n) * ld + k0 + sub * 8;
    v16bf b;
#pragma unroll
    for (int e = 0; e < 8; ++e) { b[e] = p[e]; b[e + 8] = p[e + 16]; }
    return b;
}

__device__ inline v8f zero8() {
    v8f a;
#pragma unroll
    for (int r = 0; r < 8; ++r) a[r] = 0.0f;
    return a;
}

// Parallel LayerNorm row statistics: one wave per row, lane-parallel + ds_swizzle tree.
__device__ inline void row_stats(const float* Cbuf, int ncol, int nrows, float* mu, float* rs) {
    int wave = threadIdx.x >> 5, lane = threadIdx.x & 31;
    int nwaves = blockDim.x >> 5;
    float inv = 1.0f / (float)ncol;
    for (int r = wave; r < nrows; r += nwaves) {
        float s = 0.0f, s2 = 0.0f;
        for (int c = lane; c < ncol; c += 32) {
            float v = Cbuf[r * ncol + c];
            s += v; s2 += v * v;
        }
        s = swz_add<16>(s);  s = swz_add<8>(s);  s = swz_add<4>(s);  s = swz_add<2>(s);  s = swz_add<1>(s);
        s2 = swz_add<16>(s2); s2 = swz_add<8>(s2); s2 = swz_add<4>(s2); s2 = swz_add<2>(s2); s2 = swz_add<1>(s2);
        if (lane == 0) {
            float mm = s * inv;
            mu[r] = mm;
            rs[r] = rsqrtf(s2 * inv - mm * mm + 1e-5f);
        }
    }
}

// ---------------- prep: transpose+cvt weights to bf16, dX/dt, h, b3(.)dxdt ----------------
__global__ void prep_kernel(const float* vW1, const float* vW2, const float* vW3,
                            const float* eW1, const float* eW2,
                            const float* Wv, const float* Wo,
                            const float* dW1, const float* dW2,
                            const float* path, const float* ts, const float* b3,
                            char* ws) {
    size_t gid = (size_t)blockIdx.x * blockDim.x + threadIdx.x;
    size_t str = (size_t)gridDim.x * blockDim.x;

    bf16_t* W1T = (bf16_t*)(ws + OFF_W1T);
    for (size_t t = gid; t < 256 * 128; t += str) { int n = t >> 7, k = t & 127; W1T[t] = (bf16_t)vW1[k * 256 + n]; }
    bf16_t* W2T = (bf16_t*)(ws + OFF_W2T);
    for (size_t t = gid; t < 256 * 256; t += str) { int n = t >> 8, k = t & 255; W2T[t] = (bf16_t)vW2[k * 256 + n]; }
    bf16_t* W3T = (bf16_t*)(ws + OFF_W3T);
    for (size_t t = gid; t < 4096ull * 256; t += str) { int n = (int)(t >> 8), k = (int)(t & 255); W3T[t] = (bf16_t)vW3[(size_t)k * 4096 + n]; }
    bf16_t* EW1T = (bf16_t*)(ws + OFF_EW1T);
    for (size_t t = gid; t < 128 * 32; t += str) { int n = t >> 5, k = t & 31; EW1T[t] = (bf16_t)eW1[k * 128 + n]; }
    bf16_t* EW2T = (bf16_t*)(ws + OFF_EW2T);
    for (size_t t = gid; t < 128 * 128; t += str) { int n = t >> 7, k = t & 127; EW2T[t] = (bf16_t)eW2[k * 128 + n]; }
    bf16_t* WVT = (bf16_t*)(ws + OFF_WVT);
    for (size_t t = gid; t < 128 * 128; t += str) { int n = t >> 7, k = t & 127; WVT[t] = (bf16_t)Wv[k * 128 + n]; }
    bf16_t* WOT = (bf16_t*)(ws + OFF_WOT);
    for (size_t t = gid; t < 128 * 128; t += str) { int n = t >> 7, k = t & 127; WOT[t] = (bf16_t)Wo[k * 128 + n]; }
    bf16_t* DW1T = (bf16_t*)(ws + OFF_DW1T);
    for (size_t t = gid; t < 128 * 128; t += str) { int n = t >> 7, k = t & 127; DW1T[t] = (bf16_t)dW1[k * 128 + n]; }
    bf16_t* DW2T = (bf16_t*)(ws + OFF_DW2T);
    for (size_t t = gid; t < 16 * 128; t += str) { int n = t >> 7, k = t & 127; DW2T[t] = (n < 10) ? (bf16_t)dW2[k * 10 + n] : (bf16_t)0.0f; }

    // dX/dt : [31][128][32]
    float* dXs = (float*)(ws + OFF_DXS);
    for (size_t t = gid; t < 31ull * 128 * 32; t += str) {
        int s = (int)(t / (128 * 32)); int r = (int)(t % (128 * 32));
        int b = r >> 5, i = r & 31;
        float dt = ts[b * 32 + s + 1] - ts[b * 32 + s];
        dXs[t] = (path[b * 1024 + (s + 1) * 32 + i] - path[b * 1024 + s * 32 + i]) / dt;
    }
    // h : [31][128]
    float* hb = (float*)(ws + OFF_HB);
    for (size_t t = gid; t < 31ull * 128; t += str) {
        int s = (int)(t >> 7), b = (int)(t & 127);
        hb[t] = ts[b * 32 + s + 1] - ts[b * 32 + s];
    }
    // biasK[s][b][h] = sum_i b3[h*32+i] * dxdt[b,s,i]
    float* BK = (float*)(ws + OFF_BK);
    for (size_t t = gid; t < 31ull * 128 * 128; t += str) {
        int s = (int)(t / 16384); int r = (int)(t % 16384);
        int b = r >> 7, h = r & 127;
        float dt = ts[b * 32 + s + 1] - ts[b * 32 + s];
        float acc = 0.0f;
#pragma unroll 8
        for (int i = 0; i < 32; ++i)
            acc += b3[h * 32 + i] * (path[b * 1024 + (s + 1) * 32 + i] - path[b * 1024 + s * 32 + i]);
        BK[t] = acc / dt;
    }
}

// ---------------- encoder: z0 = LN(GELU(LN(x0@W1+b1))@W2+b2) ----------------
__global__ __launch_bounds__(256) void encoder_kernel(const float* path,
        const float* eb1, const float* eg1, const float* ebt1,
        const float* eb2, const float* eg2, const float* ebt2, char* ws) {
    const bf16_t* EW1T = (const bf16_t*)(ws + OFF_EW1T);
    const bf16_t* EW2T = (const bf16_t*)(ws + OFF_EW2T);
    float* z = (float*)(ws + OFF_Z);
    __shared__ __align__(16) bf16_t Abuf[16 * 128];
    __shared__ float  Cbuf[16 * 128];
    __shared__ float  mu[16], rs[16];
    int tid = threadIdx.x, mb = blockIdx.x;
    int wave = tid >> 5, lane = tid & 31, sub = lane >> 4, ln = lane & 15;

    for (int t = tid; t < 16 * 32; t += 256) {
        int r = t >> 5, c = t & 31;
        Abuf[r * 32 + c] = (bf16_t)path[(mb * 16 + r) * 1024 + c];  // x0 = path[:,0,:]
    }
    __syncthreads();
    { // GEMM1: 16x128, K=32 (one WMMA per tile), 8 waves x 1 tile
        v8f acc = zero8();
        v16bf a = ldA(Abuf, 32, 0, 0);
        v16bf b = ldB(EW1T, 32, wave * 16, 0);
        acc = WMMA_BF16(a, b, acc);
        float bi = eb1[wave * 16 + ln];
#pragma unroll
        for (int r = 0; r < 8; ++r) Cbuf[(r + 8 * sub) * 128 + wave * 16 + ln] = acc[r] + bi;
    }
    __syncthreads();
    row_stats(Cbuf, 128, 16, mu, rs);
    __syncthreads();
    for (int t = tid; t < 16 * 128; t += 256) {
        int r = t >> 7, c = t & 127;
        Abuf[t] = (bf16_t)gelu_f((Cbuf[t] - mu[r]) * rs[r] * eg1[c] + ebt1[c]);
    }
    __syncthreads();
    { // GEMM2: K=128
        v8f acc = zero8();
#pragma unroll
        for (int kc = 0; kc < 4; ++kc) {
            v16bf a = ldA(Abuf, 128, 0, kc * 32);
            v16bf b = ldB(EW2T, 128, wave * 16, kc * 32);
            acc = WMMA_BF16(a, b, acc);
        }
        float bi = eb2[wave * 16 + ln];
#pragma unroll
        for (int r = 0; r < 8; ++r) Cbuf[(r + 8 * sub) * 128 + wave * 16 + ln] = acc[r] + bi;
    }
    __syncthreads();
    row_stats(Cbuf, 128, 16, mu, rs);
    __syncthreads();
    for (int t = tid; t < 16 * 128; t += 256) {
        int r = t >> 7, c = t & 127;
        z[(mb * 16 + r) * 128 + c] = (Cbuf[t] - mu[r]) * rs[r] * eg2[c] + ebt2[c];
    }
}

// ---------------- vf layer1: H1 = GELU(LN((z + coef*h*kprev)@W1 + b1)); init kcur = biasK ----------------
__global__ __launch_bounds__(512) void vf1_kernel(char* ws,
        const float* b1, const float* g1, const float* bt1,
        const float* kprev, float coef, float* kcur,
        const float* hb, const float* biasK) {
    const bf16_t* W1T = (const bf16_t*)(ws + OFF_W1T);
    const float* z = (const float*)(ws + OFF_Z);
    bf16_t* H1 = (bf16_t*)(ws + OFF_H1);
    __shared__ __align__(16) bf16_t Abuf[16 * 128];
    __shared__ float  Cbuf[16 * 256];
    __shared__ float  mu[16], rs[16];
    int tid = threadIdx.x, mb = blockIdx.x;
    int wave = tid >> 5, lane = tid & 31, sub = lane >> 4, ln = lane & 15;

    for (int t = tid; t < 16 * 128; t += 512) {
        int r = t >> 7, c = t & 127; int row = mb * 16 + r;
        kcur[row * 128 + c] = biasK[row * 128 + c];      // fold b3 (.) dxdt into k init
        float v = z[row * 128 + c];
        if (kprev) v += coef * hb[row] * kprev[row * 128 + c];
        Abuf[t] = (bf16_t)v;
    }
    __syncthreads();
    v8f acc = zero8();
#pragma unroll
    for (int kc = 0; kc < 4; ++kc) {
        v16bf a = ldA(Abuf, 128, 0, kc * 32);
        v16bf b = ldB(W1T, 128, wave * 16, kc * 32);
        acc = WMMA_BF16(a, b, acc);
    }
    float bi = b1[wave * 16 + ln];
#pragma unroll
    for (int r = 0; r < 8; ++r) Cbuf[(r + 8 * sub) * 256 + wave * 16 + ln] = acc[r] + bi;
    __syncthreads();
    row_stats(Cbuf, 256, 16, mu, rs);
    __syncthreads();
    for (int t = tid; t < 16 * 256; t += 512) {
        int r = t >> 8, c = t & 255;
        H1[(mb * 16 + r) * 256 + c] = (bf16_t)gelu_f((Cbuf[t] - mu[r]) * rs[r] * g1[c] + bt1[c]);
    }
}

// ---------------- vf layer2: H2 = GELU(LN(H1@W2 + b2)) ----------------
__global__ __launch_bounds__(512) void vf2_kernel(char* ws,
        const float* b2, const float* g2, const float* bt2) {
    const bf16_t* W2T = (const bf16_t*)(ws + OFF_W2T);
    const bf16_t* H1 = (const bf16_t*)(ws + OFF_H1);
    bf16_t* H2 = (bf16_t*)(ws + OFF_H2);
    __shared__ __align__(16) bf16_t Abuf[16 * 256];
    __shared__ float  Cbuf[16 * 256];
    __shared__ float  mu[16], rs[16];
    int tid = threadIdx.x, mb = blockIdx.x;
    int wave = tid >> 5, lane = tid & 31, sub = lane >> 4, ln = lane & 15;

    // async stage A block (16x256 bf16 = 8KB): one b128 per thread, no VGPR round-trip
    async_copy_b128(H1 + mb * 16 * 256 + tid * 8, Abuf + tid * 8);
    wait_async0();
    __syncthreads();
    v8f acc = zero8();
#pragma unroll
    for (int kc = 0; kc < 8; ++kc) {
        v16bf a = ldA(Abuf, 256, 0, kc * 32);
        v16bf b = ldB(W2T, 256, wave * 16, kc * 32);
        acc = WMMA_BF16(a, b, acc);
    }
    float bi = b2[wave * 16 + ln];
#pragma unroll
    for (int r = 0; r < 8; ++r) Cbuf[(r + 8 * sub) * 256 + wave * 16 + ln] = acc[r] + bi;
    __syncthreads();
    row_stats(Cbuf, 256, 16, mu, rs);
    __syncthreads();
    for (int t = tid; t < 16 * 256; t += 512) {
        int r = t >> 8, c = t & 255;
        H2[(mb * 16 + r) * 256 + c] = (bf16_t)gelu_f((Cbuf[t] - mu[r]) * rs[r] * g2[c] + bt2[c]);
    }
}

// ---------------- vf layer3 + einsum: kcur[b,h] += sum_i (H2@W3)[b,h*32+i]*dxdt[b,i] ----------------
__global__ __launch_bounds__(512) void vf3_kernel(char* ws, float* kcur, const float* dXs) {
    const bf16_t* W3T = (const bf16_t*)(ws + OFF_W3T);
    const bf16_t* H2 = (const bf16_t*)(ws + OFF_H2);
    __shared__ __align__(16) bf16_t Abuf[16 * 256];
    int tid = threadIdx.x;
    int mb = blockIdx.x & 7;     // M block (16 rows)
    int nb = blockIdx.x >> 3;    // N block (256 cols), 16 blocks
    int wave = tid >> 5, lane = tid & 31, sub = lane >> 4, ln = lane & 15;

    // async stage A block (16x256 bf16 = 8KB)
    async_copy_b128(H2 + mb * 16 * 256 + tid * 8, Abuf + tid * 8);
    int n0 = nb * 256 + wave * 16;
    __builtin_prefetch(W3T + (size_t)(n0 + ln) * 256, 0, 1);   // global_prefetch_b8 on B panel
    wait_async0();
    __syncthreads();
    v8f acc = zero8();
#pragma unroll
    for (int kc = 0; kc < 8; ++kc) {
        v16bf a = ldA(Abuf, 256, 0, kc * 32);
        v16bf b = ldB(W3T, 256, n0, kc * 32);
        acc = WMMA_BF16(a, b, acc);
    }
    int n = n0 + ln, i = n & 31, h = n >> 5;
#pragma unroll
    for (int r = 0; r < 8; ++r) {
        int row = mb * 16 + r + 8 * sub;
        float v = acc[r] * dXs[row * 32 + i];
        v = swz_add<8>(v);   // butterfly within 16-lane half: 1 ds_swizzle + 1 add per step
        v = swz_add<4>(v);
        v = swz_add<2>(v);
        v = swz_add<1>(v);
        if (ln == 0) atomicAdd(&kcur[row * 128 + h], v);
    }
}

// ---------------- RK4 combine: z += h/6 * (k1 + 2k2 + 2k3 + k4) ----------------
__global__ void zupd_kernel(char* ws, int s) {
    float* z = (float*)(ws + OFF_Z);
    const float* kb = (const float*)(ws + OFF_K);
    const float* hb = (const float*)(ws + OFF_HB) + s * 128;
    int t = blockIdx.x * blockDim.x + threadIdx.x;
    if (t < 16384) {
        int row = t >> 7;
        float h6 = hb[row] * (1.0f / 6.0f);
        z[t] += h6 * (kb[t] + 2.0f * kb[16384 + t] + 2.0f * kb[2 * 16384 + t] + kb[3 * 16384 + t]);
    }
}

// ---------------- head: softmax over singleton == 1 => attn = v. Chain 4 GEMMs through LDS. ----------------
__device__ void gemm128_inplace(bf16_t* buf, const bf16_t* WT, const float* bias, int act) {
    int tid = threadIdx.x, wave = tid >> 5, lane = tid & 31, sub = lane >> 4, ln = lane & 15;
    v8f acc[4];
#pragma unroll
    for (int tt = 0; tt < 4; ++tt) {
        int tile = wave + tt * 16; int mb = tile >> 3, nb = tile & 7;
        v8f a8 = zero8();
#pragma unroll
        for (int kc = 0; kc < 4; ++kc) {
            v16bf a = ldA(buf, 128, mb * 16, kc * 32);
            v16bf b = ldB(WT, 128, nb * 16, kc * 32);
            a8 = WMMA_BF16(a, b, a8);
        }
        acc[tt] = a8;
    }
    __syncthreads();   // everyone done reading buf
#pragma unroll
    for (int tt = 0; tt < 4; ++tt) {
        int tile = wave + tt * 16; int mb = tile >> 3, nb = tile & 7;
        float bi = bias[nb * 16 + ln];
#pragma unroll
        for (int r = 0; r < 8; ++r) {
            float v = acc[tt][r] + bi;
            if (act) v = gelu_f(v);
            buf[(mb * 16 + r + 8 * sub) * 128 + nb * 16 + ln] = (bf16_t)v;
        }
    }
    __syncthreads();
}

__global__ __launch_bounds__(512) void head_kernel(char* ws,
        const float* bv, const float* bo, const float* db1, const float* db2, float* out) {
    const bf16_t* WVT  = (const bf16_t*)(ws + OFF_WVT);
    const bf16_t* WOT  = (const bf16_t*)(ws + OFF_WOT);
    const bf16_t* DW1T = (const bf16_t*)(ws + OFF_DW1T);
    const bf16_t* DW2T = (const bf16_t*)(ws + OFF_DW2T);
    const float* z = (const float*)(ws + OFF_Z);
    __shared__ __align__(16) bf16_t buf[128 * 128];
    int tid = threadIdx.x, wave = tid >> 5, lane = tid & 31, sub = lane >> 4, ln = lane & 15;

    for (int t = tid; t < 16384; t += 512) buf[t] = (bf16_t)z[t];
    __syncthreads();
    gemm128_inplace(buf, WVT, bv, 0);    // attn = zT@Wv + bv   (softmax weights == 1)
    gemm128_inplace(buf, WOT, bo, 0);    // final = attn@Wo + bo
    gemm128_inplace(buf, DW1T, db1, 1);  // hD = gelu(final@dec_W1 + dec_b1)
    if (wave < 8) {                      // traits = hD@dec_W2 + dec_b2   (N=10, padded to 16)
        int mb = wave;
        v8f a8 = zero8();
#pragma unroll
        for (int kc = 0; kc < 4; ++kc) {
            v16bf a = ldA(buf, 128, mb * 16, kc * 32);
            v16bf b = ldB(DW2T, 128, 0, kc * 32);
            a8 = WMMA_BF16(a, b, a8);
        }
        if (ln < 10) {
#pragma unroll
            for (int r = 0; r < 8; ++r) {
                int row = mb * 16 + r + 8 * sub;
                out[row * 10 + ln] = a8[r] + db2[ln];
            }
        }
    }
    for (int t = tid; t < 128; t += 512) out[1280 + t] = 1.0f;  // attention_weights == 1
}

extern "C" void kernel_launch(void* const* d_in, const int* in_sizes, int n_in,
                              void* d_out, int out_size, void* d_ws, size_t ws_size,
                              hipStream_t stream) {
    const float* path = (const float*)d_in[0];
    const float* ts   = (const float*)d_in[1];
    const float* eW1  = (const float*)d_in[2];
    const float* eb1  = (const float*)d_in[3];
    const float* eg1  = (const float*)d_in[4];
    const float* ebt1 = (const float*)d_in[5];
    const float* eW2  = (const float*)d_in[6];
    const float* eb2  = (const float*)d_in[7];
    const float* eg2  = (const float*)d_in[8];
    const float* ebt2 = (const float*)d_in[9];
    const float* vW1  = (const float*)d_in[10];
    const float* vb1  = (const float*)d_in[11];
    const float* vg1  = (const float*)d_in[12];
    const float* vbt1 = (const float*)d_in[13];
    const float* vW2  = (const float*)d_in[14];
    const float* vb2  = (const float*)d_in[15];
    const float* vg2  = (const float*)d_in[16];
    const float* vbt2 = (const float*)d_in[17];
    const float* vW3  = (const float*)d_in[18];
    const float* vb3  = (const float*)d_in[19];
    const float* Wv   = (const float*)d_in[24];
    const float* bv   = (const float*)d_in[25];
    const float* Wo   = (const float*)d_in[26];
    const float* bo   = (const float*)d_in[27];
    const float* dW1  = (const float*)d_in[28];
    const float* db1  = (const float*)d_in[29];
    const float* dW2  = (const float*)d_in[30];
    const float* db2  = (const float*)d_in[31];
    char* ws = (char*)d_ws;
    float* out = (float*)d_out;

    prep_kernel<<<1024, 256, 0, stream>>>(vW1, vW2, vW3, eW1, eW2, Wv, Wo, dW1, dW2, path, ts, vb3, ws);
    encoder_kernel<<<8, 256, 0, stream>>>(path, eb1, eg1, ebt1, eb2, eg2, ebt2, ws);

    float* kbase = (float*)(ws + OFF_K);
    float* hbase = (float*)(ws + OFF_HB);
    float* bkbase = (float*)(ws + OFF_BK);
    float* dxbase = (float*)(ws + OFF_DXS);
    const float coefs[4] = {0.0f, 0.5f, 0.5f, 1.0f};

    for (int s = 0; s < 31; ++s) {
        const float* hb = hbase + s * 128;
        const float* biasK = bkbase + (size_t)s * 16384;
        const float* dXs = dxbase + (size_t)s * 128 * 32;
        for (int j = 0; j < 4; ++j) {
            const float* kprev = (j > 0) ? (kbase + (j - 1) * 16384) : nullptr;
            float* kcur = kbase + j * 16384;
            vf1_kernel<<<8, 512, 0, stream>>>(ws, vb1, vg1, vbt1, kprev, coefs[j], kcur, hb, biasK);
            vf2_kernel<<<8, 512, 0, stream>>>(ws, vb2, vg2, vbt2);
            vf3_kernel<<<128, 512, 0, stream>>>(ws, kcur, dXs);
        }
        zupd_kernel<<<64, 256, 0, stream>>>(ws, s);
    }
    head_kernel<<<1, 512, 0, stream>>>(ws, bv, bo, db1, db2, out);
    (void)in_sizes; (void)n_in; (void)out_size; (void)ws_size;
}
